// GraphGAT_75746043232803
// MI455X (gfx1250) — matrix operand order repaired and checked by reading, not compiled, not run on hardware
//
#include <hip/hip_runtime.h>
#include <hip/hip_bf16.h>

typedef __attribute__((ext_vector_type(16))) __bf16 v16bf;
typedef __attribute__((ext_vector_type(8)))  float  v8f;

#define GAT_ALPHA 0.2f
#define NEG_BIG  -1.0e12f
#define L2E       1.4426950408889634f   // log2(e)

// one-instruction XOR-pattern lane shuffle: ds_swizzle group-of-32 encoding
template<int XORMASK>
__device__ __forceinline__ float swz_xor(float v) {
    return __int_as_float(__builtin_amdgcn_ds_swizzle(
        __float_as_int(v), (XORMASK << 10) | 0x1F));
}

// ---------------------------------------------------------------------------
// Kernel 1: h = x @ W (bf16 WMMA, fp32 acc); emit s1 = log2e*(h@a1),
// s2 = log2e*(h@a2) (pre-scaled so kernel 2's softmax runs natively in
// base-2) and hbfT = bf16 transpose of h (64 x 8192) so kernel 2 gets
// contiguous B tiles. One wave computes one 16x64 tile of h. 512 tiles.
// ---------------------------------------------------------------------------
__global__ __launch_bounds__(256)
void gat_h_s_kernel(const float* __restrict__ x, const float* __restrict__ W,
                    const float* __restrict__ a,
                    __bf16* __restrict__ hbfT,
                    float* __restrict__ s1, float* __restrict__ s2)
{
    const int lane  = threadIdx.x & 31;
    const int wave  = threadIdx.x >> 5;
    const int tile  = blockIdx.x * 8 + wave;       // 64 blocks * 8 waves = 512
    const int m0    = tile * 16;
    const int Mrow  = m0 + (lane & 15);
    const int khalf = lane >> 4;
    const int col   = lane & 15;

    v8f acc[4] = {};
    for (int kc = 0; kc < 256; kc += 32) {
        // A operand: 16x32, lanes0-15 K=0..15, lanes16-31 K=16..31 (matches B)
        v16bf A;
        const float* xr = x + Mrow * 256 + kc + khalf * 16;
        #pragma unroll
        for (int e = 0; e < 8; ++e) A[e] = (__bf16)xr[e];
        #pragma unroll
        for (int e = 0; e < 8; ++e) A[8 + e] = (__bf16)xr[8 + e];

        #pragma unroll
        for (int t = 0; t < 4; ++t) {
            // B operand: 32x16, V0..7 lanes0-15 K=0..15, lanes16-31 K=16..31
            v16bf B;
            const float* wr = W + (kc + khalf * 16) * 64 + t * 16 + col;
            #pragma unroll
            for (int e = 0; e < 16; ++e) B[e] = (__bf16)wr[e * 64];
            acc[t] = __builtin_amdgcn_wmma_f32_16x16x32_bf16(
                false, A, false, B, (short)0, acc[t], false, false);
        }
    }

    // C layout: VGPR r, lanes0-15 -> row m0+r, lanes16-31 -> row m0+8+r
    const int base = (lane < 16) ? 0 : 8;

    float s1p[8], s2p[8];
    #pragma unroll
    for (int r = 0; r < 8; ++r) { s1p[r] = 0.f; s2p[r] = 0.f; }

    #pragma unroll
    for (int t = 0; t < 4; ++t) {
        const float a1 = a[t * 16 + col];
        const float a2 = a[64 + t * 16 + col];
        #pragma unroll
        for (int r = 0; r < 8; ++r) {
            const float v = acc[t][r];
            s1p[r] += v * a1;
            s2p[r] += v * a2;
            hbfT[(t * 16 + col) * 8192 + m0 + base + r] = (__bf16)v;
        }
    }
    // XOR-tree reduce across the 16 lanes of each half-wave (one DS op each)
    #pragma unroll
    for (int r = 0; r < 8; ++r) {
        s1p[r] += swz_xor<1>(s1p[r]);  s2p[r] += swz_xor<1>(s2p[r]);
        s1p[r] += swz_xor<2>(s1p[r]);  s2p[r] += swz_xor<2>(s2p[r]);
        s1p[r] += swz_xor<4>(s1p[r]);  s2p[r] += swz_xor<4>(s2p[r]);
        s1p[r] += swz_xor<8>(s1p[r]);  s2p[r] += swz_xor<8>(s2p[r]);
    }
    if (col == 0) {
        #pragma unroll
        for (int r = 0; r < 8; ++r) {
            s1[m0 + base + r] = s1p[r] * L2E;   // pre-scale: base-2 softmax
            s2[m0 + base + r] = s2p[r] * L2E;
        }
    }
}

// ---------------------------------------------------------------------------
// Kernel 2: streaming (flash-style) masked softmax + attention @ h.
// Block = 4 waves; all 4 own the same 16 output rows but scan disjoint
// 2048-column segments of adj (2048 independent waves machine-wide).
// Softmax tracked in base-2 (inputs pre-scaled by log2e). Partials merged
// in LDS with a base-2 log-sum-exp combine.
// ---------------------------------------------------------------------------
__global__ __launch_bounds__(128)
void gat_attn_kernel(const int* __restrict__ adj,
                     const float* __restrict__ s1, const float* __restrict__ s2,
                     const __bf16* __restrict__ hbfT,
                     float* __restrict__ out)
{
    __shared__ float sm_m[4][16];
    __shared__ float sm_l[4][16];
    __shared__ float sm_acc[4][16][64];

    const int lane  = threadIdx.x & 31;
    const int wave  = threadIdx.x >> 5;
    const int m0    = blockIdx.x * 16;             // 512 blocks
    const int M     = lane & 15;
    const int khalf = lane >> 4;
    const int base  = (lane < 16) ? 0 : 8;
    const int col   = lane & 15;
    const int row   = m0 + M;

    const float s1v = s1[row];                     // already * log2e
    const int* arow = adj + (long long)row * 8192;

    const int jbeg = wave * 2048;
    const int jend = jbeg + 2048;

    float m = -3.0e12f;     // below NEG_BIG so the first chunk always wins
    float l = 0.f;
    v8f acc[4] = {};

    for (int j0 = jbeg; j0 < jend; j0 += 32) {
        const int k1 = j0 + khalf * 16;       // K set of this lane (A layout)
        const int k2 = j0 + khalf * 16 + 8;

        const int4 q0 = *(const int4*)(arow + k1);
        const int4 q1 = *(const int4*)(arow + k1 + 4);
        const int4 q2 = *(const int4*)(arow + k2);
        const int4 q3 = *(const int4*)(arow + k2 + 4);
        const float4 t0 = *(const float4*)(s2 + k1);
        const float4 t1 = *(const float4*)(s2 + k1 + 4);
        const float4 t2 = *(const float4*)(s2 + k2);
        const float4 t3 = *(const float4*)(s2 + k2 + 4);

        int av[16]; float sv[16];
        av[0]=q0.x; av[1]=q0.y; av[2]=q0.z; av[3]=q0.w;
        av[4]=q1.x; av[5]=q1.y; av[6]=q1.z; av[7]=q1.w;
        av[8]=q2.x; av[9]=q2.y; av[10]=q2.z; av[11]=q2.w;
        av[12]=q3.x; av[13]=q3.y; av[14]=q3.z; av[15]=q3.w;
        sv[0]=t0.x; sv[1]=t0.y; sv[2]=t0.z; sv[3]=t0.w;
        sv[4]=t1.x; sv[5]=t1.y; sv[6]=t1.z; sv[7]=t1.w;
        sv[8]=t2.x; sv[9]=t2.y; sv[10]=t2.z; sv[11]=t2.w;
        sv[12]=t3.x; sv[13]=t3.y; sv[14]=t3.z; sv[15]=t3.w;

        float ev[16];   // = log2e * leaky_relu(s1+s2), or NEG_BIG if masked
        #pragma unroll
        for (int e = 0; e < 16; ++e) {
            float t = s1v + sv[e];
            t = fmaxf(t, GAT_ALPHA * t);           // scale commutes with leaky
            ev[e] = (av[e] > 0) ? t : NEG_BIG;
        }

        float cmax = ev[0];
        #pragma unroll
        for (int e = 1; e < 16; ++e) cmax = fmaxf(cmax, ev[e]);
        cmax = fmaxf(cmax, swz_xor<16>(cmax));     // join row halves (1 DS op)
        const float newm = fmaxf(m, cmax);

        float p[16], ps = 0.f;
        #pragma unroll
        for (int e = 0; e < 16; ++e) {
            p[e] = __builtin_amdgcn_exp2f(ev[e] - newm);   // bare v_exp_f32
            ps += p[e];
        }
        ps += swz_xor<16>(ps);

        if (__any(newm > m)) {
            // rare path: running max moved for at least one row -> rescale
            const float scale = __builtin_amdgcn_exp2f(m - newm);
            l = l * scale + ps;
            float sc[8];
            #pragma unroll
            for (int r = 0; r < 8; ++r) sc[r] = __shfl(scale, base + r, 32);
            #pragma unroll
            for (int t = 0; t < 4; ++t) {
                #pragma unroll
                for (int r = 0; r < 8; ++r) acc[t][r] *= sc[r];
            }
        } else {
            l += ps;
        }
        m = newm;

        // probabilities -> bf16 A operand (layout matches lane K ownership)
        v16bf A;
        #pragma unroll
        for (int e = 0; e < 16; ++e) A[e] = (__bf16)p[e];

        #pragma unroll
        for (int t = 0; t < 4; ++t) {
            const v16bf B = *(const v16bf*)(hbfT +
                (t * 16 + (lane & 15)) * 8192 + j0 + khalf * 16);
            acc[t] = __builtin_amdgcn_wmma_f32_16x16x32_bf16(
                false, A, false, B, (short)0, acc[t], false, false);
        }
    }

    // dump per-wave partials to LDS
    if (lane < 16) { sm_m[wave][M] = m; sm_l[wave][M] = l; }
    #pragma unroll
    for (int t = 0; t < 4; ++t) {
        #pragma unroll
        for (int r = 0; r < 8; ++r)
            sm_acc[wave][base + r][t * 16 + col] = acc[t][r];
    }
    __syncthreads();

    // merge the 4 segment-partials (base-2 log-sum-exp), finalize with ELU
    for (int idx = threadIdx.x; idx < 16 * 64; idx += 128) {
        const int r = idx >> 6;
        const int c = idx & 63;
        const float M4 = fmaxf(fmaxf(sm_m[0][r], sm_m[1][r]),
                               fmaxf(sm_m[2][r], sm_m[3][r]));
        float L = 0.f, v = 0.f;
        #pragma unroll
        for (int s = 0; s < 4; ++s) {
            const float f = __builtin_amdgcn_exp2f(sm_m[s][r] - M4);
            L += sm_l[s][r] * f;
            v += sm_acc[s][r][c] * f;
        }
        v /= L;
        v = (v > 0.f) ? v : (__expf(v) - 1.0f);        // elu (alpha=1)
        out[(m0 + r) * 64 + c] = v;
    }
}

extern "C" void kernel_launch(void* const* d_in, const int* in_sizes, int n_in,
                              void* d_out, int out_size, void* d_ws, size_t ws_size,
                              hipStream_t stream)
{
    const float* x   = (const float*)d_in[0];   // 8192 x 256
    const int*   adj = (const int*)  d_in[1];   // 8192 x 8192
    const float* W   = (const float*)d_in[2];   // 256 x 64
    const float* a   = (const float*)d_in[3];   // 128 x 1
    float* out = (float*)d_out;                 // 8192 x 64 fp32

    char* ws = (char*)d_ws;
    __bf16* hbfT = (__bf16*)ws;                          // 64*8192*2 = 1 MiB
    float*  s1   = (float*)(ws + 64 * 8192 * 2);         // 8192 floats
    float*  s2   = s1 + 8192;                            // 8192 floats

    gat_h_s_kernel <<<64, 256, 0, stream>>>(x, W, a, hbfT, s1, s2);
    gat_attn_kernel<<<512, 128, 0, stream>>>(adj, s1, s2, hbfT, out);
}
